// MultiHeadSelfAttention_83837761618640
// MI455X (gfx1250) — matrix operand order repaired and checked by reading, not compiled, not run on hardware
//
#include <hip/hip_runtime.h>

// ---- problem constants --------------------------------------------------
constexpr int cB  = 2;
constexpr int cS  = 2048;
constexpr int cD  = 1024;
constexpr int cH  = 16;
constexpr int cDK = 64;

constexpr int LDA = 40;   // LDS row stride (in bf16) for 32-wide K tiles (80B = 5*16B)
constexpr int LDK = 72;   // LDS row stride (in bf16) for 64-wide tiles  (144B = 9*16B)

typedef __attribute__((ext_vector_type(16))) __bf16 v16bf;
typedef __attribute__((ext_vector_type(8)))  float  v8f;
typedef __attribute__((address_space(3)))    unsigned short lds_us;

union FragBF {      // 16 bf16 = 32 bytes = two 16B chunks
    uint4 q[2];
    v16bf v;
};

__device__ __forceinline__ unsigned short f2bf(float f) {
    // native v_cvt to bf16 (RNE) instead of manual bit twiddling
    return __builtin_bit_cast(unsigned short, (__bf16)f);
}

__device__ __forceinline__ v8f wmma_bf16(v16bf a, v16bf b, v8f c) {
    // D = A(16x32 bf16) * B(32x16 bf16) + C(16x16 f32)
    return __builtin_amdgcn_wmma_f32_16x16x32_bf16(false, a, false, b,
                                                   (short)0, c, false, false);
}

// gfx1250 async copy: memory -> LDS, 16 bytes per lane, tracked by ASYNCcnt.
// GVS addressing: SGPR 64-bit base + per-lane 32-bit byte offset.
__device__ __forceinline__ void async_b128(unsigned short* lds_ptr,
                                           const unsigned short* sbase,
                                           unsigned goff_bytes) {
    lds_us* lp = (lds_us*)lds_ptr;
    asm volatile("global_load_async_to_lds_b128 %0, %1, %2"
                 :: "v"(lp), "v"(goff_bytes), "s"(sbase)
                 : "memory");
}
__device__ __forceinline__ void wait_async0() {
    asm volatile("s_wait_asynccnt 0" ::: "memory");
}

// =========================================================================
// Kernel 1: fused Q/K/V projection (x @ W^T) + RoPE(Q,K) + bf16 pack
//   grid = (D/64, (B*S)/64, 3), block = 128 (4 waves)
//   Q,K -> [B,H,S,DK] bf16 ; V -> [B,H,DK,S] bf16 (transposed for attention)
// =========================================================================
__global__ __launch_bounds__(128) void qkv_rope_kernel(
    const float* __restrict__ x,
    const float* __restrict__ Wq, const float* __restrict__ Wk,
    const float* __restrict__ Wv,
    unsigned short* __restrict__ q_bf, unsigned short* __restrict__ k_bf,
    unsigned short* __restrict__ vt_bf)
{
    __shared__ unsigned short Alds[64 * LDA];
    __shared__ unsigned short Blds[64 * LDA];

    const int mtile = blockIdx.y * 64;
    const int ntile = blockIdx.x * 64;
    const int mode  = blockIdx.z;                 // 0=Q 1=K 2=V
    const float* __restrict__ W = (mode == 0) ? Wq : ((mode == 1) ? Wk : Wv);

    const int t    = threadIdx.x;
    const int lane = t & 31;
    const int wv   = t >> 5;
    const int half = lane >> 4;
    const int l16  = lane & 15;

    v8f acc[4];
    #pragma unroll
    for (int i = 0; i < 4; ++i)
        acc[i] = v8f{0.f, 0.f, 0.f, 0.f, 0.f, 0.f, 0.f, 0.f};

    for (int k0 = 0; k0 < cD; k0 += 32) {
        __syncthreads();
        // stage x tile 64x32 f32 -> bf16 LDS
        #pragma unroll
        for (int i = 0; i < 4; ++i) {
            int idx = t + i * 128;                // 0..511 float4's
            int row = idx >> 3;
            int c4  = (idx & 7) << 2;
            float4 f = *(const float4*)(x + (size_t)(mtile + row) * cD + k0 + c4);
            unsigned short* dst = &Alds[row * LDA + c4];
            dst[0] = f2bf(f.x); dst[1] = f2bf(f.y);
            dst[2] = f2bf(f.z); dst[3] = f2bf(f.w);
        }
        // stage W tile 64x32 f32 -> bf16 LDS  (W is [N][K] row-major = W^T as B-matrix)
        #pragma unroll
        for (int i = 0; i < 4; ++i) {
            int idx = t + i * 128;
            int row = idx >> 3;
            int c4  = (idx & 7) << 2;
            float4 f = *(const float4*)(W + (size_t)(ntile + row) * cD + k0 + c4);
            unsigned short* dst = &Blds[row * LDA + c4];
            dst[0] = f2bf(f.x); dst[1] = f2bf(f.y);
            dst[2] = f2bf(f.z); dst[3] = f2bf(f.w);
        }
        __syncthreads();

        // A fragment (16x32): row = wv*16 + l16, K = 16*half + e (contiguous)
        FragBF a;
        const unsigned short* ap = &Alds[(wv * 16 + l16) * LDA];
        a.q[0] = *(const uint4*)(ap + 16 * half);
        a.q[1] = *(const uint4*)(ap + 16 * half + 8);

        #pragma unroll
        for (int ns = 0; ns < 4; ++ns) {
            // B fragment (32x16): N = ns*16 + l16, K = 16*half + e (contiguous)
            FragBF b;
            const unsigned short* bp = &Blds[(ns * 16 + l16) * LDA + 16 * half];
            b.q[0] = *(const uint4*)(bp);
            b.q[1] = *(const uint4*)(bp + 8);
            acc[ns] = wmma_bf16(a.v, b.v, acc[ns]);
        }
    }

    // ---- epilogue: RoPE + pack ----
    #pragma unroll
    for (int ns = 0; ns < 4; ++ns) {
        const int N  = ntile + ns * 16 + l16;
        const int h  = N >> 6;
        const int dk = N & 63;
        #pragma unroll
        for (int r = 0; r < 8; ++r) {
            const int M = mtile + wv * 16 + 8 * half + r;
            const int b = M >> 11;                // / 2048
            const int s = M & (cS - 1);
            float v = acc[ns][r];
            if (mode < 2) {
                const int   i2  = dk & ~1;                        // 2*(dk/2)
                const float ang = (float)s *
                    __expf(-(float)i2 * (9.210340371976184f / 64.f)); // 10000^(-2i/64)
                const float cs = cosf(ang), sn = sinf(ang);
                const float partner = __shfl_xor(v, 1, 32);        // even<->odd feature
                v = (dk & 1) ? (partner * sn + v * cs)
                             : (v * cs - partner * sn);
                const size_t o = ((size_t)(b * cH + h) * cS + s) * cDK + dk;
                ((mode == 0) ? q_bf : k_bf)[o] = f2bf(v);
            } else {
                const size_t o = ((size_t)(b * cH + h) * cDK + dk) * cS + s;
                vt_bf[o] = f2bf(v);               // V transposed: [bh][dk][s]
            }
        }
    }
}

// =========================================================================
// Kernel 2: causal flash attention, online softmax, double-buffered
//           async-to-LDS K/V staging (ASYNCcnt pipeline)
//   grid = (S/64, B*H), block = 128 (4 waves, 16 query rows each)
// =========================================================================
__global__ __launch_bounds__(128) void attn_kernel(
    const unsigned short* __restrict__ q_bf,
    const unsigned short* __restrict__ k_bf,
    const unsigned short* __restrict__ vt_bf,
    unsigned short* __restrict__ att_bf)
{
    __shared__ unsigned short Klds[2][64 * LDK];   // [key][dk], double buffered
    __shared__ unsigned short Vlds[2][64 * LDK];   // [dk][key] (V^T), double buffered
    __shared__ unsigned short Plds[64 * LDK];      // per-wave 16x64 P, bf16

    const int qb = blockIdx.x;
    const int bh = blockIdx.y;
    const int b  = bh >> 4;
    const int h  = bh & 15;

    const int t    = threadIdx.x;
    const int lane = t & 31;
    const int wv   = t >> 5;
    const int half = lane >> 4;
    const int l16  = lane & 15;

    const size_t qkb = (size_t)bh * cS * cDK;
    const size_t vtb = (size_t)bh * cDK * cS;
    const unsigned short* kbase = k_bf + qkb;
    const unsigned short* vbase = vt_bf + vtb;

    // issue async copy of key-block j into LDS buffer `sel` (8 x b128 / thread)
    auto issue = [&](int j, int sel) {
        #pragma unroll
        for (int i = 0; i < 4; ++i) {
            int idx = t + i * 128;                // 512 x 8-halves
            int row = idx >> 3;
            int c8  = (idx & 7) << 3;
            async_b128(&Klds[sel][row * LDK + c8], kbase,
                       (unsigned)(((j * 64 + row) * cDK + c8) * 2));
            async_b128(&Vlds[sel][row * LDK + c8], vbase,
                       (unsigned)((row * cS + j * 64 + c8) * 2));
        }
    };

    issue(0, 0);   // prologue: tile 0 in flight while we fetch Q

    // Q fragments, loaded once (A-layout straight from global: contiguous 16B runs)
    FragBF aq[2];
    {
        const int sq = qb * 64 + wv * 16 + l16;
        const unsigned short* qp = q_bf + qkb + (size_t)sq * cDK;
        #pragma unroll
        for (int c = 0; c < 2; ++c) {
            aq[c].q[0] = *(const uint4*)(qp + c * 32 + 16 * half);
            aq[c].q[1] = *(const uint4*)(qp + c * 32 + 16 * half + 8);
        }
    }

    v8f o[4];
    #pragma unroll
    for (int i = 0; i < 4; ++i)
        o[i] = v8f{0.f, 0.f, 0.f, 0.f, 0.f, 0.f, 0.f, 0.f};
    float m[8], l[8];
    #pragma unroll
    for (int r = 0; r < 8; ++r) { m[r] = -3.0e38f; l[r] = 0.f; }

    for (int j = 0; j <= qb; ++j) {
        wait_async0();          // this wave's tile-j copies are done
        __syncthreads();        // => everyone's copies are done
        if (j < qb) issue(j + 1, (j + 1) & 1);   // overlap next tile with compute

        const unsigned short* Kt = &Klds[j & 1][0];
        const unsigned short* Vt = &Vlds[j & 1][0];

        // ---- S = Q K^T : 2 K-chunks x 4 N-tiles ----
        v8f sa[4];
        #pragma unroll
        for (int i = 0; i < 4; ++i)
            sa[i] = v8f{0.f, 0.f, 0.f, 0.f, 0.f, 0.f, 0.f, 0.f};
        #pragma unroll
        for (int c = 0; c < 2; ++c) {
            #pragma unroll
            for (int ns = 0; ns < 4; ++ns) {
                FragBF kb;
                const unsigned short* kp =
                    &Kt[(ns * 16 + l16) * LDK + c * 32 + 16 * half];
                kb.q[0] = *(const uint4*)(kp);
                kb.q[1] = *(const uint4*)(kp + 8);
                sa[ns] = wmma_bf16(aq[c].v, kb.v, sa[ns]);
            }
        }

        // ---- online softmax update (clip +-80, /sqrt(64), causal mask) ----
        float p[4][8];
        #pragma unroll
        for (int r = 0; r < 8; ++r) {
            const int qi = qb * 64 + wv * 16 + 8 * half + r;
            float zmax = -3.0e38f;
            #pragma unroll
            for (int ns = 0; ns < 4; ++ns) {
                const int ki = j * 64 + ns * 16 + l16;
                float z = fminf(fmaxf(sa[ns][r], -80.f), 80.f) * 0.125f;
                if (ki > qi) z = -3.0e38f;
                p[ns][r] = z;
                zmax = fmaxf(zmax, z);
            }
            #pragma unroll
            for (int d = 1; d < 16; d <<= 1)       // row lives in a 16-lane half
                zmax = fmaxf(zmax, __shfl_xor(zmax, d, 32));

            const float mn   = fmaxf(m[r], zmax);
            const float corr = __expf(m[r] - mn);
            float rs = 0.f;
            #pragma unroll
            for (int ns = 0; ns < 4; ++ns) {
                float e = __expf(p[ns][r] - mn);
                p[ns][r] = e;
                rs += e;
            }
            #pragma unroll
            for (int d = 1; d < 16; d <<= 1)
                rs += __shfl_xor(rs, d, 32);
            l[r] = l[r] * corr + rs;
            m[r] = mn;
            #pragma unroll
            for (int nt = 0; nt < 4; ++nt) o[nt][r] *= corr;
            // park P (bf16) in per-wave LDS to re-shape C-layout -> A-layout
            #pragma unroll
            for (int ns = 0; ns < 4; ++ns)
                Plds[(wv * 16 + 8 * half + r) * LDK + ns * 16 + l16] =
                    f2bf(p[ns][r]);
        }

        // ---- O += P V : 2 key-chunks x 4 dk-tiles (same-wave LDS is in-order) ----
        #pragma unroll
        for (int c = 0; c < 2; ++c) {
            FragBF ap;
            const unsigned short* pp = &Plds[(wv * 16 + l16) * LDK + c * 32];
            ap.q[0] = *(const uint4*)(pp + 16 * half);
            ap.q[1] = *(const uint4*)(pp + 16 * half + 8);
            #pragma unroll
            for (int nt = 0; nt < 4; ++nt) {
                FragBF bv;
                const unsigned short* vp =
                    &Vt[(nt * 16 + l16) * LDK + c * 32 + 16 * half];
                bv.q[0] = *(const uint4*)(vp);
                bv.q[1] = *(const uint4*)(vp + 8);
                o[nt] = wmma_bf16(ap.v, bv.v, o[nt]);
            }
        }
    }

    // ---- normalize and write att in [B,S,D] bf16 ----
    #pragma unroll
    for (int r = 0; r < 8; ++r) {
        const float inv = 1.f / l[r];
        const int   s   = qb * 64 + wv * 16 + 8 * half + r;
        #pragma unroll
        for (int nt = 0; nt < 4; ++nt) {
            const int dk = nt * 16 + l16;
            att_bf[(size_t)(b * cS + s) * cD + h * cDK + dk] =
                f2bf(o[nt][r] * inv);
        }
    }
}

// =========================================================================
// Kernel 3: output projection  out = att @ Wo^T  (f32 result)
//   grid = (D/64, (B*S)/64), block = 128
// =========================================================================
__global__ __launch_bounds__(128) void oproj_kernel(
    const unsigned short* __restrict__ att_bf,
    const float* __restrict__ Wo,
    float* __restrict__ out)
{
    __shared__ unsigned short Alds[64 * LDA];
    __shared__ unsigned short Blds[64 * LDA];

    const int mtile = blockIdx.y * 64;
    const int ntile = blockIdx.x * 64;
    const int t    = threadIdx.x;
    const int lane = t & 31;
    const int wv   = t >> 5;
    const int half = lane >> 4;
    const int l16  = lane & 15;

    v8f acc[4];
    #pragma unroll
    for (int i = 0; i < 4; ++i)
        acc[i] = v8f{0.f, 0.f, 0.f, 0.f, 0.f, 0.f, 0.f, 0.f};

    for (int k0 = 0; k0 < cD; k0 += 32) {
        __syncthreads();
        // att tile is already bf16: straight b128 copies
        #pragma unroll
        for (int i = 0; i < 2; ++i) {
            int idx = t + i * 128;                // 256 x 8-halves
            int row = idx >> 2;
            int c8  = (idx & 3) << 3;
            *(uint4*)&Alds[row * LDA + c8] =
                *(const uint4*)(att_bf + (size_t)(mtile + row) * cD + k0 + c8);
        }
        // Wo tile f32 -> bf16
        #pragma unroll
        for (int i = 0; i < 4; ++i) {
            int idx = t + i * 128;
            int row = idx >> 3;
            int c4  = (idx & 7) << 2;
            float4 f = *(const float4*)(Wo + (size_t)(ntile + row) * cD + k0 + c4);
            unsigned short* dst = &Blds[row * LDA + c4];
            dst[0] = f2bf(f.x); dst[1] = f2bf(f.y);
            dst[2] = f2bf(f.z); dst[3] = f2bf(f.w);
        }
        __syncthreads();

        FragBF a;
        const unsigned short* ap = &Alds[(wv * 16 + l16) * LDA];
        a.q[0] = *(const uint4*)(ap + 16 * half);
        a.q[1] = *(const uint4*)(ap + 16 * half + 8);
        #pragma unroll
        for (int ns = 0; ns < 4; ++ns) {
            FragBF b;
            const unsigned short* bp = &Blds[(ns * 16 + l16) * LDA + 16 * half];
            b.q[0] = *(const uint4*)(bp);
            b.q[1] = *(const uint4*)(bp + 8);
            acc[ns] = wmma_bf16(a.v, b.v, acc[ns]);
        }
    }

    #pragma unroll
    for (int ns = 0; ns < 4; ++ns) {
        const int N = ntile + ns * 16 + l16;
        #pragma unroll
        for (int r = 0; r < 8; ++r) {
            const int M = mtile + wv * 16 + 8 * half + r;
            out[(size_t)M * cD + N] = acc[ns][r];
        }
    }
}

// =========================================================================
extern "C" void kernel_launch(void* const* d_in, const int* in_sizes, int n_in,
                              void* d_out, int out_size, void* d_ws, size_t ws_size,
                              hipStream_t stream)
{
    (void)in_sizes; (void)n_in; (void)out_size; (void)ws_size;
    const float* x  = (const float*)d_in[0];
    const float* Wq = (const float*)d_in[1];
    const float* Wk = (const float*)d_in[2];
    const float* Wv = (const float*)d_in[3];
    const float* Wo = (const float*)d_in[4];
    float* out = (float*)d_out;

    const size_t elems = (size_t)cB * cH * cS * cDK;   // 4,194,304 per tensor
    unsigned short* q_bf   = (unsigned short*)d_ws;
    unsigned short* k_bf   = q_bf   + elems;
    unsigned short* vt_bf  = k_bf   + elems;
    unsigned short* att_bf = vt_bf  + elems;           // total 32 MB of ws

    qkv_rope_kernel<<<dim3(cD / 64, (cB * cS) / 64, 3), 128, 0, stream>>>(
        x, Wq, Wk, Wv, q_bf, k_bf, vt_bf);
    attn_kernel<<<dim3(cS / 64, cB * cH), 128, 0, stream>>>(
        q_bf, k_bf, vt_bf, att_bf);
    oproj_kernel<<<dim3(cD / 64, (cB * cS) / 64), 128, 0, stream>>>(
        att_bf, Wo, out);
}
